// half_integer_2bit_8col_45990509806218
// MI455X (gfx1250) — matrix operand order repaired and checked by reading, not compiled, not run on hardware
//
#include <hip/hip_runtime.h>

// CDNA5 / gfx1250 nearest-codeword vector quantizer.
// scores = 2*|X| @ G^T - ||g||^2 via V_WMMA_F32_16X16X32_F16, with the -||g||^2
// bias folded into the GEMM as a 9th K element (A[k=8]=1, B[k=8]=-norm).
// Argmax epilogue: u64 lexicographic keys + DPP ROW_SHR fold (no ds_bpermute).

typedef __attribute__((ext_vector_type(16))) _Float16 v16h;
typedef __attribute__((ext_vector_type(8)))  float    v8f;

#define NTILES 15   // ceil(227/16) codeword tiles (columns padded to 240)

union VU { v16h h; unsigned u[8]; };

__device__ __forceinline__ unsigned pack_h2(float a, float b) {
    union { _Float16 f; unsigned short u; } ca, cb;
    ca.f = (_Float16)a; cb.f = (_Float16)b;
    return (unsigned)ca.u | ((unsigned)cb.u << 16);
}

// 64-bit lane-local DPP move (two 32-bit v_mov_b32_dpp), CTRL = row_shr:N etc.
template <int CTRL>
__device__ __forceinline__ unsigned long long dpp_mov_u64(unsigned long long k) {
    unsigned lo = (unsigned)__builtin_amdgcn_update_dpp(
        0, (int)(unsigned)(k & 0xFFFFFFFFull), CTRL, 0xF, 0xF, true);
    unsigned hi = (unsigned)__builtin_amdgcn_update_dpp(
        0, (int)(unsigned)(k >> 32), CTRL, 0xF, 0xF, true);
    return ((unsigned long long)hi << 32) | lo;
}

template <int CTRL>
__device__ __forceinline__ void red_step(unsigned long long& k) {
    unsigned long long o = dpp_mov_u64<CTRL>(k);
    if (o > k) k = o;           // v_cmp_lt_u64 + 2x cndmask
}

#define ROW_SHR_1 0x111
#define ROW_SHR_2 0x112
#define ROW_SHR_4 0x114
#define ROW_SHR_8 0x118

__global__ __launch_bounds__(256)
void vq_wmma_kernel(const float* __restrict__ X,
                    const float* __restrict__ grid,    // [gcount, 8]
                    const float* __restrict__ gnorm,   // [gcount]
                    float* __restrict__ out,           // [8N vals][N idx-as-float]
                    int n, int gcount) {
    // B tiles pre-packed in WMMA 16-bit B(32x16) per-lane layout:
    // slot (t, lane): lane<16 -> grid[16t+lane][0..7] f16 in dwords 0..3,
    //                 dword4 lo-half = -norm (K=8 bias), rest zero.
    __shared__ uint4 ldsB[NTILES * 32 * 2];   // 15 KiB
    __shared__ int   ldsR[8][32];             // per-wave winner redistribution

    const int tid  = threadIdx.x;
    const int lane = tid & 31;
    const int wid  = tid >> 5;

    // ---------------- prep: pack codebook (+bias) into LDS ----------------
    for (int slot = tid; slot < NTILES * 32; slot += 256) {
        int t = slot >> 5, l = slot & 31;
        uint4 lo = make_uint4(0u, 0u, 0u, 0u);
        uint4 hi = make_uint4(0u, 0u, 0u, 0u);
        if (l < 16) {
            int col = t * 16 + l;
            float nf = 65504.f;               // padded col: score ~= -65504
            if (col < gcount) {
                const float* g = grid + (size_t)col * 8;
                lo.x = pack_h2(g[0], g[1]);
                lo.y = pack_h2(g[2], g[3]);
                lo.z = pack_h2(g[4], g[5]);
                lo.w = pack_h2(g[6], g[7]);
                nf = gnorm[col];              // exact in f16 (k/4, <= 10)
            }
            hi.x = pack_h2(-nf, 0.f);         // B[K=8][col] = -||g||^2
        }
        ldsB[slot * 2 + 0] = lo;
        ldsB[slot * 2 + 1] = hi;
    }
    __syncthreads();

    const int ngroups = (n + 31) >> 5;        // 32 rows per wave-iteration
    const int gstride = gridDim.x * 8;
    int group = blockIdx.x * 8 + wid;

    const v8f cz = {};
    const unsigned ONE_H = 0x00003C00u;       // f16 1.0 in low half (A[K=8]=1)

    for (; group < ngroups; group += gstride) {
        const int row = group * 32 + lane;    // lane owns this row

        // ---- load own row (2x b128, coalesced), signs + 2|x| in f16 ----
        float4 x01 = make_float4(0.f, 0.f, 0.f, 0.f);
        float4 x23 = make_float4(0.f, 0.f, 0.f, 0.f);
        if (row < n) {
            const float4* xp = (const float4*)(X + (size_t)row * 8);
            x01 = xp[0]; x23 = xp[1];
        }
        unsigned sb =  (__float_as_uint(x01.x) >> 31)
                    | ((__float_as_uint(x01.y) >> 31) << 1)
                    | ((__float_as_uint(x01.z) >> 31) << 2)
                    | ((__float_as_uint(x01.w) >> 31) << 3)
                    | ((__float_as_uint(x23.x) >> 31) << 4)
                    | ((__float_as_uint(x23.y) >> 31) << 5)
                    | ((__float_as_uint(x23.z) >> 31) << 6)
                    | ((__float_as_uint(x23.w) >> 31) << 7);
        unsigned p0 = pack_h2(2.f * fabsf(x01.x), 2.f * fabsf(x01.y));
        unsigned p1 = pack_h2(2.f * fabsf(x01.z), 2.f * fabsf(x01.w));
        unsigned p2 = pack_h2(2.f * fabsf(x23.x), 2.f * fabsf(x23.y));
        unsigned p3 = pack_h2(2.f * fabsf(x23.z), 2.f * fabsf(x23.w));

        // A tiles: 16-bit A 16x32 layout — lanes 0-15: K0..7 in dwords 0..3,
        // dword4 lo-half = 1.0 bias element (matches B); lanes 16-31 zero.
        const bool lo16 = lane < 16;
        VU a0u, a1u;
        a0u.u[0] = lo16 ? p0 : 0u;
        a0u.u[1] = lo16 ? p1 : 0u;
        a0u.u[2] = lo16 ? p2 : 0u;
        a0u.u[3] = lo16 ? p3 : 0u;
        a0u.u[4] = lo16 ? ONE_H : 0u; a0u.u[5] = 0u; a0u.u[6] = 0u; a0u.u[7] = 0u;
        unsigned q0 = __shfl_xor(p0, 16, 32);
        unsigned q1 = __shfl_xor(p1, 16, 32);
        unsigned q2 = __shfl_xor(p2, 16, 32);
        unsigned q3 = __shfl_xor(p3, 16, 32);
        a1u.u[0] = lo16 ? q0 : 0u;
        a1u.u[1] = lo16 ? q1 : 0u;
        a1u.u[2] = lo16 ? q2 : 0u;
        a1u.u[3] = lo16 ? q3 : 0u;
        a1u.u[4] = lo16 ? ONE_H : 0u; a1u.u[5] = 0u; a1u.u[6] = 0u; a1u.u[7] = 0u;

        float best0[8], best1[8];
        int   bidx0[8], bidx1[8];
#pragma unroll
        for (int v = 0; v < 8; ++v) {
            best0[v] = -3.0e38f; best1[v] = -3.0e38f;
            bidx0[v] = 0;        bidx1[v] = 0;
        }

        const int colbase = lane & 15;

        // ------------- score tiles + running argmax (scores pre-biased) -------------
#pragma unroll 5
        for (int t = 0; t < NTILES; ++t) {
            VU bu;
            uint4 b0 = ldsB[(t * 32 + lane) * 2 + 0];
            uint4 b1 = ldsB[(t * 32 + lane) * 2 + 1];
            bu.u[0] = b0.x; bu.u[1] = b0.y; bu.u[2] = b0.z; bu.u[3] = b0.w;
            bu.u[4] = b1.x; bu.u[5] = b1.y; bu.u[6] = b1.z; bu.u[7] = b1.w;

            v8f d0 = __builtin_amdgcn_wmma_f32_16x16x32_f16(
                false, a0u.h, false, bu.h, (short)0, cz, false, false);
            v8f d1 = __builtin_amdgcn_wmma_f32_16x16x32_f16(
                false, a1u.h, false, bu.h, (short)0, cz, false, false);

            const int col = t * 16 + colbase;
#pragma unroll
            for (int v = 0; v < 8; ++v) {
                // strict > keeps the lowest column on ties within this lane
                if (d0[v] > best0[v]) { best0[v] = d0[v]; bidx0[v] = col; }
                if (d1[v] > best1[v]) { best1[v] = d1[v]; bidx1[v] = col; }
            }
        }

        // ---- exact lexicographic keys: {sortable(score) , 255-col} ----
        unsigned long long k0[8], k1[8];
#pragma unroll
        for (int v = 0; v < 8; ++v) {
            unsigned u0 = __float_as_uint(best0[v]);
            unsigned m0 = u0 ^ ((unsigned)((int)u0 >> 31) | 0x80000000u);
            k0[v] = ((unsigned long long)m0 << 32) | (unsigned)(255 - bidx0[v]);
            unsigned u1 = __float_as_uint(best1[v]);
            unsigned m1 = u1 ^ ((unsigned)((int)u1 >> 31) | 0x80000000u);
            k1[v] = ((unsigned long long)m1 << 32) | (unsigned)(255 - bidx1[v]);
        }

        // ---- DPP ROW_SHR fold: lanes 0 and 16 end with the per-half argmax ----
#pragma unroll
        for (int v = 0; v < 8; ++v) {
            red_step<ROW_SHR_8>(k0[v]);
            red_step<ROW_SHR_4>(k0[v]);
            red_step<ROW_SHR_2>(k0[v]);
            red_step<ROW_SHR_1>(k0[v]);
            red_step<ROW_SHR_8>(k1[v]);
            red_step<ROW_SHR_4>(k1[v]);
            red_step<ROW_SHR_2>(k1[v]);
            red_step<ROW_SHR_1>(k1[v]);
        }

        // ---- redistribute 32 winners to owning lanes via per-wave LDS ----
        // group0 half0 -> rows 0..7, half1 -> 8..15; group1 -> rows 16..31.
        if ((lane & 15) == 0) {
            int h8 = (lane >> 4) * 8;
#pragma unroll
            for (int v = 0; v < 8; ++v) {
                ldsR[wid][h8 + v]      = (int)(k0[v] & 0xFFu);
                ldsR[wid][16 + h8 + v] = (int)(k1[v] & 0xFFu);
            }
        }
        __builtin_amdgcn_wave_barrier();   // same-wave DS ops are in-order
        int widx = 255 - ldsR[wid][lane];
        __builtin_amdgcn_wave_barrier();

        // ---------------- epilogue: gather codeword, sign, store ----------------
        if (row < n) {
            int id16 = (int)(sb << 8) + widx - 32768;   // exact in int16 range
            out[(size_t)n * 8 + row] = (float)id16;

            const float4* gp = (const float4*)(grid + (size_t)widx * 8);
            float4 g0 = gp[0], g1 = gp[1];
            g0.x = (sb & 1u)   ? -g0.x : g0.x;
            g0.y = (sb & 2u)   ? -g0.y : g0.y;
            g0.z = (sb & 4u)   ? -g0.z : g0.z;
            g0.w = (sb & 8u)   ? -g0.w : g0.w;
            g1.x = (sb & 16u)  ? -g1.x : g1.x;
            g1.y = (sb & 32u)  ? -g1.y : g1.y;
            g1.z = (sb & 64u)  ? -g1.z : g1.z;
            g1.w = (sb & 128u) ? -g1.w : g1.w;
            float4* ov = (float4*)(out + (size_t)row * 8);
            ov[0] = g0; ov[1] = g1;
        }
    }
}

extern "C" void kernel_launch(void* const* d_in, const int* in_sizes, int n_in,
                              void* d_out, int out_size, void* d_ws, size_t ws_size,
                              hipStream_t stream) {
    const float* X     = (const float*)d_in[0];
    const float* grid  = (const float*)d_in[1];
    const float* gnorm = (const float*)d_in[2];
    // d_in[3] (int_map = 2^k) is implicit in the bit packing.
    const int n      = in_sizes[0] / 8;
    const int gcount = in_sizes[1] / 8;

    const int blocks = 512;   // 4096 waves, grid-stride over 32-row groups
    vq_wmma_kernel<<<blocks, 256, 0, stream>>>(X, grid, gnorm, (float*)d_out,
                                               n, gcount);
}
